// GraphEncoder_29016799052365
// MI455X (gfx1250) — compile-verified, matrix operand order checked by reading, not run on hardware
//
#include <hip/hip_runtime.h>
#include <hip/hip_bf16.h>

#define N_NODES  500000
#define N_EDGES  600000
#define N_GRAPHS 4096
#define X_DIM    128
#define H_DIM    128
#define Z_DIM    64
#define BN_EPS   1e-5f

typedef __bf16 bf16_t;
typedef bf16_t v16bf __attribute__((ext_vector_type(16)));
typedef float  v8f   __attribute__((ext_vector_type(8)));
typedef unsigned int v4u __attribute__((ext_vector_type(4)));
typedef int v8i __attribute__((ext_vector_type(8)));
typedef int v4i __attribute__((ext_vector_type(4)));

// native bf16 convert (backend RNE fptrunc -> hw cvt on gfx1250)
__device__ __forceinline__ bf16_t f2bf(float f) { return (bf16_t)f; }

// ---------------- kernel 0: zero msg + gsum + cnt -----------------
__global__ void k_zero(float4* p, int n4) {
  int i = blockIdx.x * blockDim.x + threadIdx.x;
  int stride = gridDim.x * blockDim.x;
  float4 z = {0.f, 0.f, 0.f, 0.f};
  for (; i < n4; i += stride) p[i] = z;
}

// ---------------- kernel 1: edge scatter (one wave32 per edge) ----
__global__ void k_edge_scatter(const float* __restrict__ x,
                               const int* __restrict__ ei,
                               const float* __restrict__ ew,
                               float* __restrict__ msg) {
  int wid  = (blockIdx.x * blockDim.x + threadIdx.x) >> 5;
  int lane = threadIdx.x & 31;
  if (wid >= N_EDGES) return;
  int src = ei[wid];
  int dst = ei[N_EDGES + wid];
  float w = ew[wid];
  const float4* xs = (const float4*)(x + (size_t)src * X_DIM);
  float* md = msg + (size_t)dst * X_DIM;
  float4 v = xs[lane];
  atomicAdd(md + lane * 4 + 0, w * v.x);
  atomicAdd(md + lane * 4 + 1, w * v.y);
  atomicAdd(md + lane * 4 + 2, w * v.z);
  atomicAdd(md + lane * 4 + 3, w * v.w);
}

// ---------------- kernel 2: node counts per graph -----------------
__global__ void k_count(const int* __restrict__ batch, float* __restrict__ cnt) {
  int i = blockIdx.x * blockDim.x + threadIdx.x;
  if (i < N_NODES) atomicAdd(&cnt[batch[i]], 1.0f);
}

// ---------------- kernel 3: pack W_self/W_nbr into WMMA B layout --
// packed index: ((w*8 + nt)*4 + kb)*32 + lane, each entry v16bf.
// B-operand (K=32 x N=16, bf16): lane l -> n = l%16; element e ->
// k = kb*32 + (e&7) + 16*(e>>3) + 8*(l>>4)
__global__ void k_pack_w(const float* __restrict__ Wself,
                         const float* __restrict__ Wnbr,
                         v16bf* __restrict__ packed) {
  int idx = blockIdx.x * blockDim.x + threadIdx.x;
  if (idx >= 2048) return;
  int lane = idx & 31;
  int kb   = (idx >> 5) & 3;
  int nt   = (idx >> 7) & 7;
  int w    = idx >> 10;
  const float* W = w ? Wnbr : Wself;
  int n = nt * 16 + (lane & 15);
  int half = lane >> 4;
  v16bf v;
#pragma unroll
  for (int e = 0; e < 16; ++e) {
    int k = kb * 32 + (e & 7) + 16 * (e >> 3) + 8 * half;
    v[e] = f2bf(W[k * H_DIM + n]);
  }
  packed[idx] = v;
}

// ---------------- kernel 4: fused WMMA GEMM + bias + relu + pool --
// 10 waves per block; wave 0 TDM-loads the 64KB packed-weight block
// into LDS (tensor_load_to_lds + s_wait_tensorcnt), then each wave
// computes a 16-node x 128 tile with v_wmma_f32_16x16x32_bf16 reading
// B operands from LDS, and pools relu(acc+b) into per-graph sums.
__global__ void __launch_bounds__(320)
k_node_gemm_pool(const float* __restrict__ x,
                 const float* __restrict__ msg,
                 const v16bf* __restrict__ packed,
                 const float* __restrict__ bemb,
                 const int* __restrict__ batch,
                 float* __restrict__ gsum) {
  __shared__ v16bf lds_w[2048];   // 64 KB: [(w*8+nt)*4+kb]*32 + lane

  // ---- TDM: DMA packed weights (65536 B) into LDS, wave 0 only ----
  if (threadIdx.x < 32) {
    unsigned long long ga = (unsigned long long)(const void*)packed;
    unsigned lds_off = (unsigned)(size_t)(void*)lds_w;
    // D# group0: count=1 | lds_addr | global_addr[56:0] | type=2
    v4u g0 = { 1u,
               lds_off,
               (unsigned)(ga & 0xffffffffull),
               (unsigned)((ga >> 32) & 0x1ffffffull) | (2u << 30) };
    // D# group1: data_size=8B; tensor_dim0=8192 (@bit48); tensor_dim1=1
    // (@bit80); tile_dim0=8192 (@bit112); tile_dim1=1 (@bit128);
    // tensor_dim0_stride=8192 (@bit160)
    v8i g1 = { 0x30000, 0x20000000, 0x10000, 0x20000000,
               1, 0x2000, 0, 0 };
    v4i g2 = { 0, 0, 0, 0 };
    v4i g3 = { 0, 0, 0, 0 };
    v8i g4 = { 0, 0, 0, 0, 0, 0, 0, 0 };
    __builtin_amdgcn_tensor_load_to_lds(g0, g1, g2, g3, g4, 0);
    __builtin_amdgcn_s_wait_tensorcnt(0);
  }
  __syncthreads();

  int wave = blockIdx.x * 10 + ((int)threadIdx.x >> 5);   // exact: 3125*10 = 31250
  int lane = threadIdx.x & 31;
  int m    = lane & 15;
  int half = lane >> 4;
  size_t row = (size_t)wave * 16 + m;
  const float* xr = x   + row * X_DIM;
  const float* mr = msg + row * X_DIM;

  v8f acc[8] = {};

#pragma unroll
  for (int kb = 0; kb < 4; ++kb) {
    int k0 = kb * 32 + half * 8;
    v16bf ax, am;
#pragma unroll
    for (int e = 0; e < 8; ++e) {
      ax[e]     = f2bf(xr[k0 + e]);
      ax[e + 8] = f2bf(xr[k0 + 16 + e]);
      am[e]     = f2bf(mr[k0 + e]);
      am[e + 8] = f2bf(mr[k0 + 16 + e]);
    }
#pragma unroll
    for (int nt = 0; nt < 8; ++nt) {
      v16bf bx = lds_w[(nt * 4 + kb) * 32 + lane];
      v16bf bm = lds_w[((8 + nt) * 4 + kb) * 32 + lane];
      acc[nt] = __builtin_amdgcn_wmma_f32_16x16x32_bf16(
          false, ax, false, bx, (short)0, acc[nt], false, false);
      acc[nt] = __builtin_amdgcn_wmma_f32_16x16x32_bf16(
          false, am, false, bm, (short)0, acc[nt], false, false);
    }
  }

  // C/D layout: lanes 0-15 -> M = r, N = lane; lanes 16-31 -> M = r+8.
  // batch is sorted: if the whole 16-node tile is one graph (common case),
  // pre-reduce the 8 rows per lane -> 1 atomic instead of 8.
  int gfirst = batch[(size_t)wave * 16];
  int glast  = batch[(size_t)wave * 16 + 15];
  if (gfirst == glast) {
#pragma unroll
    for (int nt = 0; nt < 8; ++nt) {
      int col = nt * 16 + m;
      float b = bemb[col];
      float s = 0.f;
#pragma unroll
      for (int r = 0; r < 8; ++r) s += fmaxf(acc[nt][r] + b, 0.0f);
      atomicAdd(&gsum[(size_t)gfirst * H_DIM + col], s);
    }
  } else {
#pragma unroll
    for (int nt = 0; nt < 8; ++nt) {
      int col = nt * 16 + m;
      float b = bemb[col];
#pragma unroll
      for (int r = 0; r < 8; ++r) {
        size_t node = (size_t)wave * 16 + r + 8 * half;
        int g = batch[node];
        float v = fmaxf(acc[nt][r] + b, 0.0f);
        atomicAdd(&gsum[(size_t)g * H_DIM + col], v);
      }
    }
  }
}

// ---------------- kernel 5: tiny head GEMMs [4096x129]x[129x64] ---
__global__ void k_heads(const float* __restrict__ gsum,
                        const float* __restrict__ cnt,
                        const float* __restrict__ y,
                        const float* __restrict__ Wmu, const float* __restrict__ bmu,
                        const float* __restrict__ Wvar, const float* __restrict__ bvar,
                        float* __restrict__ a_mu, float* __restrict__ a_var) {
  int g = blockIdx.x;
  __shared__ float semb[H_DIM + 1];
  int t = threadIdx.x;                  // 128 threads
  float c = fmaxf(cnt[g], 1.0f);
  semb[t] = gsum[(size_t)g * H_DIM + t] / c;
  if (t == 0) semb[H_DIM] = y[g];
  __syncthreads();
  int head = t >> 6, z = t & 63;
  const float* W = head ? Wvar : Wmu;
  float s = head ? bvar[z] : bmu[z];
  for (int k = 0; k < H_DIM + 1; ++k) s += semb[k] * W[k * Z_DIM + z];
  (head ? a_var : a_mu)[(size_t)g * Z_DIM + z] = s;
}

// ---------------- kernel 6: batchnorm + relu (+sigmoid) -----------
__global__ void k_bn(const float* __restrict__ a_mu, const float* __restrict__ a_var,
                     const float* __restrict__ gamma_mu, const float* __restrict__ beta_mu,
                     const float* __restrict__ gamma_var, const float* __restrict__ beta_var,
                     float* __restrict__ out) {
  int head = blockIdx.x >> 6;
  int z    = blockIdx.x & 63;
  const float* a = head ? a_var : a_mu;
  int t = threadIdx.x;                  // 256 threads
  float s = 0.f, s2 = 0.f;
  for (int g = t; g < N_GRAPHS; g += 256) {
    float v = a[(size_t)g * Z_DIM + z];
    s += v; s2 += v * v;
  }
  __shared__ float rs[256], rq[256];
  rs[t] = s; rq[t] = s2;
  __syncthreads();
  for (int o = 128; o > 0; o >>= 1) {
    if (t < o) { rs[t] += rs[t + o]; rq[t] += rq[t + o]; }
    __syncthreads();
  }
  float mean = rs[0] * (1.0f / N_GRAPHS);
  float var  = rq[0] * (1.0f / N_GRAPHS) - mean * mean;
  float inv  = rsqrtf(var + BN_EPS);
  float gm = (head ? gamma_var : gamma_mu)[z];
  float bt = (head ? beta_var  : beta_mu)[z];
  float* o = out + (head ? (size_t)N_GRAPHS * Z_DIM : 0);
  for (int g = t; g < N_GRAPHS; g += 256) {
    float v = (a[(size_t)g * Z_DIM + z] - mean) * inv * gm + bt;
    v = fmaxf(v, 0.0f);
    if (head) v = 1.0f / (1.0f + __expf(-v));
    o[(size_t)g * Z_DIM + z] = v;
  }
}

extern "C" void kernel_launch(void* const* d_in, const int* in_sizes, int n_in,
                              void* d_out, int out_size, void* d_ws, size_t ws_size,
                              hipStream_t stream) {
  const float* x         = (const float*)d_in[0];
  const int*   ei        = (const int*)  d_in[1];
  const float* ew        = (const float*)d_in[2];
  const float* y_target  = (const float*)d_in[3];
  const int*   batch     = (const int*)  d_in[4];
  const float* W_self    = (const float*)d_in[5];
  const float* W_nbr     = (const float*)d_in[6];
  const float* b_emb     = (const float*)d_in[7];
  const float* W_mu      = (const float*)d_in[8];
  const float* b_mu      = (const float*)d_in[9];
  const float* gamma_mu  = (const float*)d_in[10];
  const float* beta_mu   = (const float*)d_in[11];
  const float* W_var     = (const float*)d_in[12];
  const float* b_var     = (const float*)d_in[13];
  const float* gamma_var = (const float*)d_in[14];
  const float* beta_var  = (const float*)d_in[15];
  float* out = (float*)d_out;

  // workspace layout (floats)
  float* msg  = (float*)d_ws;                         // 500000*128
  float* gsum = msg  + (size_t)N_NODES * H_DIM;       // 4096*128
  float* cnt  = gsum + (size_t)N_GRAPHS * H_DIM;      // 4096
  float* amu  = cnt  + N_GRAPHS;                      // 4096*64
  float* avar = amu  + (size_t)N_GRAPHS * Z_DIM;      // 4096*64
  v16bf* packed = (v16bf*)(avar + (size_t)N_GRAPHS * Z_DIM); // 2048 * 32B

  // 0) zero msg + gsum + cnt
  int zero_f = N_NODES * H_DIM + N_GRAPHS * H_DIM + N_GRAPHS;
  k_zero<<<4096, 256, 0, stream>>>((float4*)d_ws, zero_f / 4);

  // 1) pack weights into WMMA B layout
  k_pack_w<<<8, 256, 0, stream>>>(W_self, W_nbr, packed);

  // 2) edge scatter
  k_edge_scatter<<<(N_EDGES * 32 + 255) / 256, 256, 0, stream>>>(x, ei, ew, msg);

  // 3) counts
  k_count<<<(N_NODES + 255) / 256, 256, 0, stream>>>(batch, cnt);

  // 4) fused WMMA GEMM + relu + pool  (31250 tiles, 10 waves/block)
  k_node_gemm_pool<<<3125, 320, 0, stream>>>(x, msg, packed, b_emb, batch, gsum);

  // 5) head GEMMs
  k_heads<<<N_GRAPHS, 128, 0, stream>>>(gsum, cnt, y_target, W_mu, b_mu,
                                        W_var, b_var, amu, avar);

  // 6) batchnorm + activations -> d_out
  k_bn<<<2 * Z_DIM, 256, 0, stream>>>(amu, avar, gamma_mu, beta_mu,
                                      gamma_var, beta_var, out);
}
